// Mnist_lmdSplineKAN_22771916603476
// MI455X (gfx1250) — matrix operand order repaired and checked by reading, not compiled, not run on hardware
//
#include <hip/hip_runtime.h>
#include <hip/hip_bf16.h>
#include <math.h>

typedef __attribute__((ext_vector_type(16))) _Float16 v16h;
typedef __attribute__((ext_vector_type(8)))  _Float16 v8h;
typedef __attribute__((ext_vector_type(8)))  float    v8f;

#define B_DIM   1024
#define I_DIM   784
#define O_DIM   64
#define H_DIM   10
#define NB      8            // spline basis count
#define K_SPL   (I_DIM * NB) // 6272
#define K_BASE  I_DIM        // 784
#define K_TOT   7072         // 6272 + 784 + 16 pad = 221 * 32
#define N_TOT   (H_DIM * O_DIM) // 640
#define HGRID   0.2f

// ---------------------------------------------------------------------------
// Kernel 1: build f16 activation matrix A[B][K_TOT]
//   cols [i*8+k]   : B-spline basis k of input i  (Cox-de Boor, K=3, uniform)
//   cols [6272+i]  : silu(x)
//   cols [7056..]  : zero pad
// ---------------------------------------------------------------------------
__global__ __launch_bounds__(256) void kan_build_A(const float* __restrict__ x,
                                                   _Float16* __restrict__ A) {
    int idx = blockIdx.x * blockDim.x + threadIdx.x;     // b*784 + i
    if (idx >= B_DIM * I_DIM) return;
    int i = idx % I_DIM;
    float xv = x[idx];

    // order-0 indicators on extended knots t_m = (m-3)*h, m = 0..11
    float bb[11];
#pragma unroll
    for (int m = 0; m < 11; ++m) {
        float t0 = (m - 3) * HGRID;
        float t1 = (m - 2) * HGRID;
        bb[m] = (xv >= t0 && xv < t1) ? 1.0f : 0.0f;
    }
    // Cox-de Boor; uniform knots -> denominators are j*h
#pragma unroll
    for (int j = 1; j <= 3; ++j) {
        float inv = 1.0f / (j * HGRID);
#pragma unroll
        for (int m = 0; m + j < 11; ++m) {
            float tm   = (m - 3) * HGRID;
            float tmj1 = (m + j - 2) * HGRID;   // t[m+j+1] = (m+j+1-3)*h
            bb[m] = ((xv - tm) * bb[m] + (tmj1 - xv) * bb[m + 1]) * inv;
        }
    }

    _Float16* arow = A + (size_t)(idx / I_DIM) * K_TOT;
#pragma unroll
    for (int k = 0; k < NB; ++k) arow[i * NB + k] = (_Float16)bb[k];
    float s = xv / (1.0f + __expf(-xv));                 // silu residual path
    arow[K_SPL + i] = (_Float16)s;
    if (i < 16) arow[K_SPL + K_BASE + i] = (_Float16)0.0f;  // pad cols
}

// ---------------------------------------------------------------------------
// Kernel 2: build f16 effective weights Weff[n = h*64+o][K_TOT]
//   cols [i*8+k]  : coef[h,i,o,k] * scale_sp[h,i,o] * lmd[h,i]
//   cols [6272+i] : scale_base[h,i,o] * lmd[h,i]
// ---------------------------------------------------------------------------
__global__ __launch_bounds__(256) void kan_build_W(const float* __restrict__ coef,
                                                   const float* __restrict__ scale_base,
                                                   const float* __restrict__ scale_sp,
                                                   const float* __restrict__ lmd,
                                                   _Float16* __restrict__ Wf) {
    int idx = blockIdx.x * blockDim.x + threadIdx.x;     // (h*784 + i)*64 + o
    if (idx >= H_DIM * I_DIM * O_DIM) return;
    int o  = idx & 63;
    int hi = idx >> 6;          // h*784 + i
    int i  = hi % I_DIM;
    int h  = hi / I_DIM;

    float l   = lmd[hi];
    float ssp = scale_sp[idx] * l;

    _Float16* wrow = Wf + (size_t)(h * O_DIM + o) * K_TOT;
    const float* cp = coef + (size_t)idx * NB;
#pragma unroll
    for (int k = 0; k < NB; ++k) wrow[i * NB + k] = (_Float16)(cp[k] * ssp);
    wrow[K_SPL + i] = (_Float16)(scale_base[idx] * l);
    if (i < 16) wrow[K_SPL + K_BASE + i] = (_Float16)0.0f;
}

// ---------------------------------------------------------------------------
// Kernel 3: WMMA GEMM  h1 = tanh(A @ Weff^T),  A: (1024 x 7072) f16,
//   Weff: (640 x 7072) f16 row-major (K contiguous), h1: (1024 x 640) f32.
// One wave computes a 16x64 tile: A fragment reused across 4 v_wmma.
// Fragment packing used by this kernel, same k order for A and B (wave32):
//   A (16x32 f16): lane l holds M=l&15, K = (l>>4)*16 + {0..15}  (contiguous)
//   B (32x16 f16): lane l holds N=l&15, K = (l>>4)*16 + {0..15}  (contiguous)
//   C (16x16 f32): lane l, elem v -> M=(l>>4)*8+v, N=l&15
// ---------------------------------------------------------------------------
__global__ __launch_bounds__(128) void kan_wmma_gemm(const _Float16* __restrict__ A,
                                                     const _Float16* __restrict__ W,
                                                     float* __restrict__ H1) {
    const int lane  = threadIdx.x & 31;
    const int wave  = threadIdx.x >> 5;
    const int mtile = blockIdx.x * 4 + wave;     // 0..63  (1024/16 row tiles)
    const int nbase = blockIdx.y * 64;           // 0..576 (640/64 col groups)
    const int lrow  = lane & 15;
    const int half  = lane >> 4;

    const _Float16* Arow = A + (size_t)(mtile * 16 + lrow) * K_TOT + half * 16;
    const _Float16* Wrow[4];
#pragma unroll
    for (int nt = 0; nt < 4; ++nt)
        Wrow[nt] = W + (size_t)(nbase + nt * 16 + lrow) * K_TOT + half * 16;

    v8f acc[4] = {v8f{}, v8f{}, v8f{}, v8f{}};

    for (int kk = 0; kk < K_TOT; kk += 32) {
        __builtin_prefetch(Arow + kk + 256, 0, 0);       // global_prefetch_b8
        v8h alo = *(const v8h*)(Arow + kk);
        v8h ahi = *(const v8h*)(Arow + kk + 8);
        v16h a  = __builtin_shufflevector(alo, ahi,
                       0, 1, 2, 3, 4, 5, 6, 7, 8, 9, 10, 11, 12, 13, 14, 15);
#pragma unroll
        for (int nt = 0; nt < 4; ++nt) {
            v16h b = *(const v16h*)(Wrow[nt] + kk);       // 32B contiguous
            acc[nt] = __builtin_amdgcn_wmma_f32_16x16x32_f16(
                /*neg_a=*/false, a, /*neg_b=*/false, b,
                /*c_mod=*/(short)0, acc[nt],
                /*reuse_a=*/false, /*reuse_b=*/false);
        }
    }

    // epilogue: tanh -> h1
#pragma unroll
    for (int nt = 0; nt < 4; ++nt) {
        const int n = nbase + nt * 16 + lrow;
#pragma unroll
        for (int v = 0; v < 8; ++v) {
            const int m = mtile * 16 + half * 8 + v;
            H1[(size_t)m * N_TOT + n] = tanhf(acc[nt][v]);
        }
    }
}

// ---------------------------------------------------------------------------
// Kernel 4: per-(b,h) head MLP: logits = tanh(h1 @ W1 + b1) @ W2 + b2
// One wave per (b,h); lane = hidden unit d (32 of them); shuffle-reduce.
// ---------------------------------------------------------------------------
__global__ __launch_bounds__(256) void kan_head_mlp(const float* __restrict__ H1,
                                                    const float* __restrict__ W1,
                                                    const float* __restrict__ b1,
                                                    const float* __restrict__ W2,
                                                    const float* __restrict__ b2,
                                                    float* __restrict__ out) {
    int gw   = (blockIdx.x * blockDim.x + threadIdx.x) >> 5;   // wave id
    int lane = threadIdx.x & 31;
    if (gw >= B_DIM * H_DIM) return;
    int b = gw / H_DIM;
    int h = gw % H_DIM;

    const float* hrow = H1 + (size_t)b * N_TOT + h * O_DIM;
    const float* w1   = W1 + (size_t)h * O_DIM * 32;

    float s = b1[h * 32 + lane];
#pragma unroll 8
    for (int o = 0; o < O_DIM; ++o)
        s = fmaf(hrow[o], w1[o * 32 + lane], s);

    float r = tanhf(s) * W2[h * 32 + lane];
#pragma unroll
    for (int off = 16; off > 0; off >>= 1)
        r += __shfl_down(r, off, 32);

    if (lane == 0) out[b * H_DIM + h] = r + b2[h];
}

// ---------------------------------------------------------------------------
// launcher
// ---------------------------------------------------------------------------
extern "C" void kernel_launch(void* const* d_in, const int* in_sizes, int n_in,
                              void* d_out, int out_size, void* d_ws, size_t ws_size,
                              hipStream_t stream) {
    const float* x          = (const float*)d_in[0];
    const float* coef       = (const float*)d_in[1];
    const float* scale_base = (const float*)d_in[2];
    const float* scale_sp   = (const float*)d_in[3];
    const float* lmd        = (const float*)d_in[4];
    const float* W1         = (const float*)d_in[5];
    const float* b1         = (const float*)d_in[6];
    const float* W2         = (const float*)d_in[7];
    const float* b2         = (const float*)d_in[8];
    float* out = (float*)d_out;

    // workspace layout (hipMalloc base is 256B aligned; all offsets 256B aligned)
    const size_t szA = (size_t)B_DIM * K_TOT * sizeof(_Float16);   // 14,483,456
    const size_t szW = (size_t)N_TOT * K_TOT * sizeof(_Float16);   //  9,052,160
    char* ws = (char*)d_ws;
    _Float16* Af  = (_Float16*)(ws);
    _Float16* Wf  = (_Float16*)(ws + szA);
    float*    H1  = (float*)(ws + szA + szW);

    kan_build_A<<<(B_DIM * I_DIM + 255) / 256, 256, 0, stream>>>(x, Af);
    kan_build_W<<<(H_DIM * I_DIM * O_DIM + 255) / 256, 256, 0, stream>>>(
        coef, scale_base, scale_sp, lmd, Wf);

    dim3 ggrid(B_DIM / 16 / 4, N_TOT / 64);      // (16, 10), 4 waves/block
    kan_wmma_gemm<<<ggrid, 128, 0, stream>>>(Af, Wf, H1);

    int nwaves = B_DIM * H_DIM;                  // 10240 waves, 8 per block
    kan_head_mlp<<<(nwaves * 32 + 255) / 256, 256, 0, stream>>>(
        H1, W1, b1, W2, b2, out);
}
